// Attention_2327872274934
// MI455X (gfx1250) — matrix operand order repaired and checked by reading, not compiled, not run on hardware
//
#include <hip/hip_runtime.h>
#include <hip/hip_bf16.h>

// ---------------------------------------------------------------------------
// Multi-head attention (B=8, S=1024, E=1024, H=16, D=64) for gfx1250 (CDNA5)
// Engine: v_wmma_f32_16x16x32_f16 (f32 accumulate), wave32, 256-thread blocks.
// Round 3: preload all 4 B operands per K-step, then issue 4 WMMAs
//          back-to-back -> removes WMMA->VALU WAR hazard v_nops.
// ---------------------------------------------------------------------------

typedef __attribute__((ext_vector_type(16))) _Float16 v16h;
typedef __attribute__((ext_vector_type(8)))  float    v8f;

#define B_  8
#define S_  1024
#define E_  1024
#define H_  16
#define D_  64

__device__ __forceinline__ v8f wmma_f16(v16h a, v16h b, v8f c) {
  // (neg_a, A, neg_b, B, c_mod, C, reuse_a, reuse_b)
  return __builtin_amdgcn_wmma_f32_16x16x32_f16(false, a, false, b,
                                                (short)0, c, false, false);
}

// 16 contiguous f16 -> B operand (lane covers K = g*16 + 0..15)
__device__ __forceinline__ v16h ld16h(const _Float16* p) {
  union { uint4 u[2]; v16h h; } cv;
  cv.u[0] = *reinterpret_cast<const uint4*>(p);
  cv.u[1] = *reinterpret_cast<const uint4*>(p + 8);
  return cv.h;
}

// A operand from two 8-half runs (K = g*16+0..7 and g*16+8..15)
__device__ __forceinline__ v16h ld16h_split(const _Float16* p0, const _Float16* p1) {
  union { uint4 u[2]; v16h h; } cv;
  cv.u[0] = *reinterpret_cast<const uint4*>(p0);
  cv.u[1] = *reinterpret_cast<const uint4*>(p1);
  return cv.h;
}

// Build f16 operand from 16 f32 values (4x float4)
__device__ __forceinline__ v16h cvt16(float4 f0, float4 f1, float4 f2, float4 f3) {
  v16h a;
  a[0]  = (_Float16)f0.x; a[1]  = (_Float16)f0.y; a[2]  = (_Float16)f0.z; a[3]  = (_Float16)f0.w;
  a[4]  = (_Float16)f1.x; a[5]  = (_Float16)f1.y; a[6]  = (_Float16)f1.z; a[7]  = (_Float16)f1.w;
  a[8]  = (_Float16)f2.x; a[9]  = (_Float16)f2.y; a[10] = (_Float16)f2.z; a[11] = (_Float16)f2.w;
  a[12] = (_Float16)f3.x; a[13] = (_Float16)f3.y; a[14] = (_Float16)f3.z; a[15] = (_Float16)f3.w;
  return a;
}

// Load 16 consecutive f32 and convert to one f16 operand
__device__ __forceinline__ v16h ldcvt16f(const float* p) {
  float4 f0 = *(const float4*)(p);
  float4 f1 = *(const float4*)(p + 4);
  float4 f2 = *(const float4*)(p + 8);
  float4 f3 = *(const float4*)(p + 12);
  return cvt16(f0, f1, f2, f3);
}

// ---------------------------------------------------------------------------
// Kernel 1: projections  qx = q@Wq^T + bq ,  kx = k@Wk^T + bk   (f32 -> f16)
// One wave = one 16x64 strip (4 N-tiles, shared A). 16384 wave-tasks total.
//   proj==0 -> kx (also writes kxT [B,H,D,S]);  proj==1 -> qx
// ---------------------------------------------------------------------------
__global__ __launch_bounds__(256) void proj_kernel(
    const float* __restrict__ qin, const float* __restrict__ kin,
    const float* __restrict__ wq_w, const float* __restrict__ wq_b,
    const float* __restrict__ wk_w, const float* __restrict__ wk_b,
    _Float16* __restrict__ qx, _Float16* __restrict__ kx,
    _Float16* __restrict__ kxT)
{
  const int tid  = threadIdx.x;
  const int wv   = tid >> 5;
  const int lane = tid & 31;
  const int g    = lane >> 4;   // lane-group: selects K halves
  const int lr   = lane & 15;   // A row / B column within tile

  const int taskIdx  = blockIdx.x * 8 + wv;
  const int proj     = taskIdx >> 13;        // 8192 tasks per projection
  const int rem      = taskIdx & 8191;
  const int rowTile  = rem >> 4;             // 0..511 (16-row strips of 8192)
  const int colGroup = rem & 15;             // 0..15  (64-col strips of 1024)

  const float* X  = proj ? qin  : kin;
  const float* W  = proj ? wq_w : wk_w;
  const float* Bv = proj ? wq_b : wk_b;

  const int arow = rowTile * 16 + lr;        // A row this lane feeds
  const float* xr = X + (size_t)arow * E_;
  const float* wr[4];
#pragma unroll
  for (int t = 0; t < 4; ++t)
    wr[t] = W + (size_t)(colGroup * 64 + t * 16 + lr) * E_;

  v8f acc[4] = {{}, {}, {}, {}};
  for (int kb = 0; kb < E_; kb += 32) {
    __builtin_prefetch(xr + kb + 64, 0, 0);  // global_prefetch_b8
    // A: K = kb + g*16 + j, j = 0..15  (loaded once, used 4x)
    float4 fa0 = *(const float4*)(xr + kb + g * 16);
    float4 fa1 = *(const float4*)(xr + kb + g * 16 + 4);
    float4 fa2 = *(const float4*)(xr + kb + g * 16 + 8);
    float4 fa3 = *(const float4*)(xr + kb + g * 16 + 12);
    const v16h a = cvt16(fa0, fa1, fa2, fa3);
    // B: B[k][n] = W[n][k]; lane n holds K = kb + g*16 + 0..15.
    // Preload all 4 tiles into distinct registers, then WMMA burst.
    v16h bop[4];
#pragma unroll
    for (int t = 0; t < 4; ++t) bop[t] = ldcvt16f(wr[t] + kb + g * 16);
#pragma unroll
    for (int t = 0; t < 4; ++t) acc[t] = wmma_f16(a, bop[t], acc[t]);
  }

  _Float16* dst = proj ? qx : kx;
  const int batch = (rowTile * 16) >> 10;    // 16-row strip never spans batches
  const int s0    = ((rowTile * 16) & 1023) + 8 * g;

#pragma unroll
  for (int t = 0; t < 4; ++t) {
    const int wcol = colGroup * 64 + t * 16 + lr;   // output column (h*64+d)
    const float bias = Bv[wcol];
    _Float16 hv[8];
#pragma unroll
    for (int r = 0; r < 8; ++r) hv[r] = (_Float16)(acc[t][r] + bias);

    // Standard layout [B,S,H,D] == [8192,1024]; row M = rowTile*16 + r + 8g
#pragma unroll
    for (int r = 0; r < 8; ++r) {
      const int grow = rowTile * 16 + r + 8 * g;
      dst[(size_t)grow * (H_ * D_) + wcol] = hv[r];
    }

    if (!proj) {
      // Transposed copy kxT[b][h][d][s]: 8 consecutive s -> one b128 store
      const int h = wcol >> 6;
      const int d = wcol & 63;
      union { uint4 u; _Float16 h8[8]; } cv;
#pragma unroll
      for (int r = 0; r < 8; ++r) cv.h8[r] = hv[r];
      const size_t base = (((size_t)batch * H_ + h) * D_ + d) * S_ + s0;
      *reinterpret_cast<uint4*>(kxT + base) = cv.u;
    }
  }
}

// ---------------------------------------------------------------------------
// Kernel 2: score = softmax( qx . kx^T / 8 )  -> score_hb region of d_out
// Block = (b, h, 16 q-rows). 8 waves x 128 key-cols each. Strip in LDS.
// ---------------------------------------------------------------------------
__global__ __launch_bounds__(256) void score_softmax_kernel(
    const _Float16* __restrict__ qx, const _Float16* __restrict__ kx,
    float* __restrict__ score)
{
  extern __shared__ float smem[];
  float* sS   = smem;                 // [16][1024] raw scores
  float* sRed = smem + 16 * 1024;     // [16][16] partial reductions
  float* sMax = sRed + 256;           // [16]
  float* sInv = sMax + 16;            // [16]

  const int tid  = threadIdx.x;
  const int wv   = tid >> 5;
  const int lane = tid & 31;
  const int g    = lane >> 4;
  const int lr   = lane & 15;

  const int qTile = blockIdx.x & 63;
  const int h     = (blockIdx.x >> 6) & 15;
  const int b     = blockIdx.x >> 10;
  const int q0    = qTile * 16;

  // A operands (d = 0..31 and 32..63), reused across 8 column tiles
  const _Float16* qb = qx + ((size_t)(b * S_ + q0 + lr)) * (H_ * D_) + h * D_;
  const v16h a0 = ld16h_split(qb + g * 16,      qb + g * 16 + 8);
  const v16h a1 = ld16h_split(qb + 32 + g * 16, qb + 40 + g * 16);

#pragma unroll
  for (int ct = 0; ct < 8; ++ct) {
    const int col0 = wv * 128 + ct * 16;
    const _Float16* kvp =
        kx + ((size_t)(b * S_ + col0 + lr)) * (H_ * D_) + h * D_;
    const v16h b0 = ld16h(kvp + g * 16);        // d = g*16 + 0..15
    const v16h b1 = ld16h(kvp + 32 + g * 16);   // d = 32 + g*16 + 0..15
    v8f acc = {};
    acc = wmma_f16(a0, b0, acc);
    acc = wmma_f16(a1, b1, acc);
#pragma unroll
    for (int r = 0; r < 8; ++r)
      sS[(r + 8 * g) * 1024 + col0 + lr] = acc[r] * 0.125f;  // 1/sqrt(64)
  }
  __syncthreads();

  // Row-wise softmax statistics: 16 threads per row
  const int row = tid >> 4;
  const int sub = tid & 15;
  float m = -3.402823466e38f;
  for (int c = sub; c < 1024; c += 16) m = fmaxf(m, sS[row * 1024 + c]);
  sRed[row * 16 + sub] = m;
  __syncthreads();
  if (sub == 0) {
    float mm = sRed[row * 16];
#pragma unroll
    for (int j = 1; j < 16; ++j) mm = fmaxf(mm, sRed[row * 16 + j]);
    sMax[row] = mm;
  }
  __syncthreads();
  const float mx = sMax[row];
  float ssum = 0.0f;
  for (int c = sub; c < 1024; c += 16)
    ssum += __expf(sS[row * 1024 + c] - mx);
  sRed[row * 16 + sub] = ssum;
  __syncthreads();
  if (sub == 0) {
    float t = 0.0f;
#pragma unroll
    for (int j = 0; j < 16; ++j) t += sRed[row * 16 + j];
    sInv[row] = 1.0f / t;
  }
  __syncthreads();

  // Coalesced normalized store: score_hb[(h*8+b)][q][k]
  const size_t obase = ((size_t)(h * B_ + b) * S_ + q0) * S_;
  for (int it = 0; it < 64; ++it) {
    const int flat = it * 256 + tid;
    const int r = flat >> 10;
    const int c = flat & 1023;
    const float v = __expf(sS[r * 1024 + c] - sMax[r]) * sInv[r];
    score[obase + (size_t)r * S_ + c] = v;
  }
}

// ---------------------------------------------------------------------------
// Kernel 3: out[b,q,h*64+d] = P . V  with P from d_out (f32->f16), V = kxT
// One wave = one 16-row M-tile x full head dim (4 N-tiles, shared A).
// ---------------------------------------------------------------------------
__global__ __launch_bounds__(256) void out_kernel(
    const float* __restrict__ score, const _Float16* __restrict__ kxT,
    float* __restrict__ out)
{
  const int tid  = threadIdx.x;
  const int wv   = tid >> 5;
  const int lane = tid & 31;
  const int g    = lane >> 4;
  const int lr   = lane & 15;

  const int mgroup = blockIdx.x & 7;
  const int h      = (blockIdx.x >> 3) & 15;
  const int b      = blockIdx.x >> 7;
  const int mtile  = mgroup * 8 + wv;            // 0..63

  const int qrow = mtile * 16 + lr;              // A row (query)
  const float* srow =
      score + ((size_t)(h * B_ + b) * S_ + qrow) * S_;
  const _Float16* vcol[4];
#pragma unroll
  for (int t = 0; t < 4; ++t)
    vcol[t] = kxT + (((size_t)b * H_ + h) * D_ + t * 16 + lr) * S_;

  v8f acc[4] = {{}, {}, {}, {}};
  for (int kb = 0; kb < S_; kb += 32) {
    __builtin_prefetch(srow + kb + 64, 0, 0);
    // A: probabilities, loaded + converted once per K-step, used 4x
    float4 f0 = *(const float4*)(srow + kb + g * 16);
    float4 f1 = *(const float4*)(srow + kb + g * 16 + 4);
    float4 f2 = *(const float4*)(srow + kb + g * 16 + 8);
    float4 f3 = *(const float4*)(srow + kb + g * 16 + 12);
    const v16h a = cvt16(f0, f1, f2, f3);
    // Preload all 4 V tiles, then WMMA burst (distinct B registers)
    v16h bop[4];
#pragma unroll
    for (int t = 0; t < 4; ++t) bop[t] = ld16h(vcol[t] + kb + g * 16);
#pragma unroll
    for (int t = 0; t < 4; ++t) acc[t] = wmma_f16(a, bop[t], acc[t]);
  }

#pragma unroll
  for (int t = 0; t < 4; ++t) {
#pragma unroll
    for (int r = 0; r < 8; ++r) {
      const int qq = mtile * 16 + r + 8 * g;
      out[((size_t)(b * S_ + qq)) * (H_ * D_) + h * D_ + t * 16 + lr] =
          acc[t][r];
    }
  }
}

// ---------------------------------------------------------------------------
extern "C" void kernel_launch(void* const* d_in, const int* in_sizes, int n_in,
                              void* d_out, int out_size, void* d_ws, size_t ws_size,
                              hipStream_t stream) {
  const float* kin  = (const float*)d_in[0];
  const float* qin  = (const float*)d_in[1];
  const float* wq_w = (const float*)d_in[2];
  const float* wq_b = (const float*)d_in[3];
  const float* wk_w = (const float*)d_in[4];
  const float* wk_b = (const float*)d_in[5];

  float* out   = (float*)d_out;
  float* score = out + (size_t)B_ * S_ * (H_ * D_);   // score_hb region

  _Float16* qx  = (_Float16*)d_ws;                    // 16 MB
  _Float16* kx  = qx + (size_t)B_ * S_ * (H_ * D_);   // 16 MB
  _Float16* kxT = kx + (size_t)B_ * S_ * (H_ * D_);   // 16 MB

  // 16384 projection wave-tasks / 8 waves per block
  proj_kernel<<<2048, 256, 0, stream>>>(qin, kin, wq_w, wq_b, wk_w, wk_b,
                                        qx, kx, kxT);

  // B*H*(S/16) = 8192 blocks; dynamic LDS: 16*1024 + 256 + 32 floats
  const size_t lds_bytes = (16 * 1024 + 256 + 32) * sizeof(float);
  score_softmax_kernel<<<8192, 256, lds_bytes, stream>>>(qx, kx, score);

  // B*H*8 m-groups = 1024 blocks
  out_kernel<<<1024, 256, 0, stream>>>(score, kxT, out);
}
